// TransolverPPBlock_14121852469799
// MI455X (gfx1250) — matrix operand order repaired and checked by reading, not compiled, not run on hardware
//
#include <hip/hip_runtime.h>
#include <math.h>

typedef _Float16 half_t;
typedef __attribute__((ext_vector_type(16))) _Float16 v16h;
typedef __attribute__((ext_vector_type(8)))  _Float16 v8h;
typedef __attribute__((ext_vector_type(8)))  float    v8f;

#define Bc 4
#define Nc 32768
#define Dc 256
#define Sc 64
#define Hc 8
#define DHc 32

// ---------------------------------------------------------------------------
// CDNA5 WMMA 16x16x32 f16 fragment layouts (wave32):
// A (16x32): m = lane&15 ; element e -> k = e + 8*((e>>3) + (lane>>4))
//            => two contiguous runs of 8 halves at kb+8*hi and kb+16+8*hi
// B (32x16): n = lane&15 ; element e -> k = e + 8*((e>>3) + (lane>>4))
//            => two contiguous runs of 8 halves at kb+8*hi and kb+16+8*hi (needs [N][K])
// C (16x16 f32): vgpr r -> m = r + 8*(lane>>4) ; n = lane&15
// ---------------------------------------------------------------------------

__device__ __forceinline__ v8f wmma_f16(v16h a, v16h b, v8f c) {
  return __builtin_amdgcn_wmma_f32_16x16x32_f16(false, a, false, b, (short)0, c,
                                                false, false);
}

// rowk = &A[m*lda + kb]  (A row-major [M][K], f16)
__device__ __forceinline__ v16h frag_a_f16(const half_t* rowk, int lane) {
  const half_t* p = rowk + ((lane >> 4) << 3);
  v8h lo = *(const v8h*)p;
  v8h hi = *(const v8h*)(p + 16);
  return __builtin_shufflevector(lo, hi, 0, 1, 2, 3, 4, 5, 6, 7, 8, 9, 10, 11,
                                 12, 13, 14, 15);
}

// colk = &Bt[n*ldk + kb]  (B stored transposed [N][K], f16)
__device__ __forceinline__ v16h frag_b_f16(const half_t* colk, int lane) {
  const half_t* p = colk + ((lane >> 4) << 3);
  v8h lo = *(const v8h*)p;
  v8h hi = *(const v8h*)(p + 16);
  return __builtin_shufflevector(lo, hi, 0, 1, 2, 3, 4, 5, 6, 7, 8, 9, 10, 11,
                                 12, 13, 14, 15);
}

// rowk = &A[m*lda + kb]  (A row-major [M][K], f32 in global; convert to f16)
__device__ __forceinline__ v16h frag_a_f32(const float* rowk, int lane) {
  const float* p = rowk + ((lane >> 4) << 3);
  v8f lo = *(const v8f*)p;
  v8f hi = *(const v8f*)(p + 16);
  v8h l = __builtin_convertvector(lo, v8h);
  v8h h = __builtin_convertvector(hi, v8h);
  return __builtin_shufflevector(l, h, 0, 1, 2, 3, 4, 5, 6, 7, 8, 9, 10, 11,
                                 12, 13, 14, 15);
}

// ---------------------------------------------------------------------------
// Generic 64-row GEMM: C[64][N] = A16[64][K] @ Wt16^T + bias
// A row-major f16 [64][K]; Wt16 is f16 stored [N][K]; all global (L2-hot).
// ---------------------------------------------------------------------------
__device__ void gemm64(const half_t* __restrict__ A, int K,
                       const half_t* __restrict__ Wt, int N,
                       const float* __restrict__ bias, float* __restrict__ C) {
  const int tid = threadIdx.x, wv = tid >> 5, lane = tid & 31;
  const int ntilesN = N >> 4;
  const int ntiles = 4 * ntilesN;  // M = 64 -> 4 M-tiles
  for (int t = wv; t < ntiles; t += 8) {
    const int mt = t / ntilesN, nt = t % ntilesN;
    v8f acc;
#pragma unroll
    for (int r = 0; r < 8; r++) acc[r] = 0.f;
    const int m = mt * 16 + (lane & 15);
    const int n = nt * 16 + (lane & 15);
    const half_t* arow = A + (size_t)m * K;
    const half_t* bcol = Wt + (size_t)n * K;
    for (int kb = 0; kb < K; kb += 32) {
      v16h a = frag_a_f16(arow + kb, lane);
      v16h b = frag_b_f16(bcol + kb, lane);
      acc = wmma_f16(a, b, acc);
    }
    const float bv = bias[n];
#pragma unroll
    for (int r = 0; r < 8; r++) {
      const int mm = mt * 16 + r + 8 * (lane >> 4);
      C[(size_t)mm * N + n] = acc[r] + bv;
    }
  }
}

// ---------------------------------------------------------------------------
// Helper kernels: zero, f32->f16 copy, f32->f16 transpose ([K][N] -> [N][K])
// ---------------------------------------------------------------------------
__global__ void k_zero(float* p, int n) {
  int i = blockIdx.x * blockDim.x + threadIdx.x;
  if (i < n) p[i] = 0.f;
}
__global__ void k_cvt(const float* __restrict__ W, half_t* __restrict__ o, int n) {
  int i = blockIdx.x * blockDim.x + threadIdx.x;
  if (i < n) o[i] = (half_t)W[i];
}
__global__ void k_cvtT(const float* __restrict__ W, half_t* __restrict__ o,
                       int K, int N) {
  int i = blockIdx.x * blockDim.x + threadIdx.x;
  if (i < K * N) {
    int k = i / N, n = i % N;
    o[(size_t)n * K + k] = (half_t)W[i];
  }
}

// ---------------------------------------------------------------------------
// Kernel 1: slice weights (softmax) + pooling.  grid(64, B), block 256.
// 16 chunks of 32 tokens per block; pool accumulates in registers; one
// atomicAdd flush per block.  Static LDS = 60 KB.
// ---------------------------------------------------------------------------
__global__ void k_pool(const float* __restrict__ x,
                       const float* __restrict__ sq_w,
                       const float* __restrict__ sq_b,
                       half_t* __restrict__ Wm16, float* __restrict__ stF) {
  __shared__ alignas(16) half_t lds_wT[Sc * Dc];    // 32 KB  sq_w^T  [s][d]
  __shared__ alignas(16) half_t lds_xT[Dc * 32];    // 16 KB  x^T    [d][tok]
  __shared__ float lds_lg[32 * Sc];                 //  8 KB  logits [tok][s]
  __shared__ alignas(16) half_t lds_wmT[Sc * 32];   //  4 KB  Wm^T   [s][tok]

  const int b = blockIdx.y;
  const int tid = threadIdx.x, wv = tid >> 5, lane = tid & 31;

  // stage sq_w transposed: lds_wT[s][d] = sq_w[d][s]
  for (int i = tid; i < Dc * Sc; i += 256) {
    int d = i >> 6, s = i & 63;
    lds_wT[s * Dc + d] = (half_t)sq_w[i];
  }

  v8f acc[8];
#pragma unroll
  for (int j = 0; j < 8; j++)
#pragma unroll
    for (int r = 0; r < 8; r++) acc[j][r] = 0.f;

  const float* xb = x + (size_t)b * Nc * Dc;

  for (int cc = 0; cc < 16; cc++) {
    const int n0 = (blockIdx.x * 16 + cc) * 32;
    __syncthreads();  // protect lds_xT / lds_wmT readers from previous iter
    // stage x chunk transposed: lds_xT[d][t] = x[n0+t][d]
    for (int i = tid; i < 32 * Dc; i += 256) {
      int t = i >> 8, d = i & 255;
      lds_xT[d * 32 + t] = (half_t)xb[(size_t)n0 * Dc + i];
    }
    // prefetch next chunk (one 128B line per thread)
    if (cc + 1 < 16)
      __builtin_prefetch(xb + (size_t)(n0 + 32) * Dc + tid * 32, 0, 1);
    __syncthreads();

    // logits GEMM: M=32, N=64, K=256 -> 8 tiles, one per wave.
    // A-frags straight from global f32 x (L2-hot), B from lds_wT.
    {
      const int mt = wv >> 2, nt = wv & 3;
      v8f c;
#pragma unroll
      for (int r = 0; r < 8; r++) c[r] = 0.f;
      const int m = mt * 16 + (lane & 15);
      const int n = nt * 16 + (lane & 15);
      const float* arow = xb + (size_t)(n0 + m) * Dc;
      const half_t* bcol = lds_wT + n * Dc;
      for (int kb = 0; kb < Dc; kb += 32) {
        v16h a = frag_a_f32(arow + kb, lane);
        v16h bfr = frag_b_f16(bcol + kb, lane);
        c = wmma_f16(a, bfr, c);
      }
      const float bv = sq_b[n];
#pragma unroll
      for (int r = 0; r < 8; r++) {
        const int mm = mt * 16 + r + 8 * (lane >> 4);
        lds_lg[mm * Sc + n] = c[r] + bv;
      }
    }
    __syncthreads();

    // softmax over S=64 per token row (threads 0..31)
    if (tid < 32) {
      const int row = tid;
      float mx = -1e30f;
      for (int s = 0; s < Sc; s++) mx = fmaxf(mx, lds_lg[row * Sc + s]);
      float sum = 0.f;
      for (int s = 0; s < Sc; s++) sum += __expf(lds_lg[row * Sc + s] - mx);
      const float inv = 1.0f / sum;
      half_t* gout = Wm16 + ((size_t)b * Nc + n0 + row) * Sc;
      for (int s = 0; s < Sc; s++) {
        float p = __expf(lds_lg[row * Sc + s] - mx) * inv;
        lds_wmT[s * 32 + row] = (half_t)p;  // transposed for pool A
        gout[s] = (half_t)p;                // row-major for unpool A
      }
    }
    __syncthreads();

    // pool GEMM accumulate: C[S=64][D=256] += Wm^T(64x32) @ x(32x256)
    // A = lds_wmT (row-major [s][tok]), B = lds_xT ([d][tok] = [N][K])
#pragma unroll
    for (int j = 0; j < 8; j++) {
      const int t = wv + 8 * j;
      const int mt = t >> 4, nt = t & 15;
      const int m = mt * 16 + (lane & 15);
      const int n = nt * 16 + (lane & 15);
      v16h a = frag_a_f16(lds_wmT + m * 32, lane);
      v16h bfr = frag_b_f16(lds_xT + n * 32, lane);
      acc[j] = wmma_f16(a, bfr, acc[j]);
    }
  }

  // flush partial pooled state
  float* stb = stF + (size_t)b * Sc * Dc;
#pragma unroll
  for (int j = 0; j < 8; j++) {
    const int t = wv + 8 * j;
    const int mt = t >> 4, nt = t & 15;
    const int n = nt * 16 + (lane & 15);
#pragma unroll
    for (int r = 0; r < 8; r++) {
      const int mm = mt * 16 + r + 8 * (lane >> 4);
      atomicAdd(&stb[mm * Dc + n], acc[j][r]);
    }
  }
}

// ---------------------------------------------------------------------------
// Kernel 2: slice-token transformer block. grid(B), block 256 (8 waves).
// All projections through WMMA gemm64 with pre-converted f16 [N][K] weights.
// ---------------------------------------------------------------------------
__global__ void k_slice(float* __restrict__ stF, half_t* __restrict__ st16,
                        float* __restrict__ qkvF, half_t* __restrict__ ctx16,
                        half_t* __restrict__ h16, float* __restrict__ tmpH,
                        float* __restrict__ tmpA,
                        const half_t* __restrict__ g_wT, const float* __restrict__ g_b,
                        const half_t* __restrict__ in_wT, const float* __restrict__ in_b,
                        const half_t* __restrict__ out_wT, const float* __restrict__ out_b,
                        const float* __restrict__ n1_s, const float* __restrict__ n1_b,
                        const half_t* __restrict__ f1_wT, const float* __restrict__ f1_b,
                        const half_t* __restrict__ f2_wT, const float* __restrict__ f2_b,
                        const float* __restrict__ n2_s, const float* __restrict__ n2_b,
                        const float* __restrict__ state) {
  __shared__ float lq[Sc * DHc];
  __shared__ float lk[Sc * DHc];
  __shared__ float lv[Sc * DHc];
  __shared__ float lsc[Sc * Sc];

  const int b = blockIdx.x;
  const int tid = threadIdx.x;

  float*  st  = stF  + (size_t)b * Sc * Dc;
  half_t* s16 = st16 + (size_t)b * Sc * Dc;
  float*  qkv = qkvF + (size_t)b * Sc * 3 * Dc;
  half_t* cx  = ctx16 + (size_t)b * Sc * Dc;
  half_t* hh  = h16  + (size_t)b * Sc * 4 * Dc;
  float*  th  = tmpH + (size_t)b * Sc * 4 * Dc;
  float*  ta  = tmpA + (size_t)b * Sc * Dc;

  for (int i = tid; i < Sc * Dc; i += 256) s16[i] = (half_t)st[i];
  __syncthreads();

  // gate = sigmoid(st @ g_w + g_b); st += gate * state
  gemm64(s16, Dc, g_wT, Dc, g_b, ta);
  __syncthreads();
  for (int i = tid; i < Sc * Dc; i += 256) {
    float g = 1.0f / (1.0f + __expf(-ta[i]));
    float v = st[i] + g * state[i];
    st[i] = v;
    s16[i] = (half_t)v;
  }
  __syncthreads();

  // qkv = st @ in_w^T + in_b
  gemm64(s16, Dc, in_wT, 3 * Dc, in_b, qkv);
  __syncthreads();

  // attention over S=64 tokens, per head (VALU; ~2 MFLOP/batch)
  const float scale = 0.1767766953f;  // 1/sqrt(32)
  for (int h = 0; h < Hc; h++) {
    __syncthreads();
    for (int i = tid; i < Sc * DHc; i += 256) {
      const int s = i / DHc, c = i % DHc;
      lq[i] = qkv[(size_t)s * 3 * Dc + h * DHc + c];
      lk[i] = qkv[(size_t)s * 3 * Dc + Dc + h * DHc + c];
      lv[i] = qkv[(size_t)s * 3 * Dc + 2 * Dc + h * DHc + c];
    }
    __syncthreads();
    for (int i = tid; i < Sc * Sc; i += 256) {
      const int r = i >> 6, c = i & 63;
      float d = 0.f;
      for (int k = 0; k < DHc; k++) d += lq[r * DHc + k] * lk[c * DHc + k];
      lsc[i] = d * scale;
    }
    __syncthreads();
    if (tid < Sc) {
      const int r = tid;
      float mx = -1e30f;
      for (int c = 0; c < Sc; c++) mx = fmaxf(mx, lsc[r * Sc + c]);
      float sum = 0.f;
      for (int c = 0; c < Sc; c++) sum += __expf(lsc[r * Sc + c] - mx);
      const float inv = 1.0f / sum;
      for (int c = 0; c < Sc; c++)
        lsc[r * Sc + c] = __expf(lsc[r * Sc + c] - mx) * inv;
    }
    __syncthreads();
    for (int i = tid; i < Sc * DHc; i += 256) {
      const int s = i >> 5, c = i & 31;
      float d = 0.f;
      for (int k = 0; k < Sc; k++) d += lsc[s * Sc + k] * lv[k * DHc + c];
      cx[(size_t)s * Dc + h * DHc + c] = (half_t)d;
    }
  }
  __syncthreads();

  // attn_out = ctx @ out_w^T + out_b ; st = LN1(st + attn_out)
  gemm64(cx, Dc, out_wT, Dc, out_b, ta);
  __syncthreads();
  if (tid < Sc) {
    const int r = tid;
    float m = 0.f;
    for (int d = 0; d < Dc; d++) m += st[r * Dc + d] + ta[r * Dc + d];
    m *= (1.0f / Dc);
    float v = 0.f;
    for (int d = 0; d < Dc; d++) {
      float y = st[r * Dc + d] + ta[r * Dc + d] - m;
      v += y * y;
    }
    const float inv = rsqrtf(v * (1.0f / Dc) + 1e-5f);
    for (int d = 0; d < Dc; d++) {
      float y = st[r * Dc + d] + ta[r * Dc + d];
      float o = (y - m) * inv * n1_s[d] + n1_b[d];
      st[r * Dc + d] = o;
      s16[r * Dc + d] = (half_t)o;
    }
  }
  __syncthreads();

  // FFN
  gemm64(s16, Dc, f1_wT, 4 * Dc, f1_b, th);
  __syncthreads();
  for (int i = tid; i < Sc * 4 * Dc; i += 256) {
    float xv = th[i];
    hh[i] = (half_t)(0.5f * xv * (1.0f + erff(xv * 0.70710678f)));
  }
  __syncthreads();
  gemm64(hh, 4 * Dc, f2_wT, Dc, f2_b, ta);
  __syncthreads();
  if (tid < Sc) {
    const int r = tid;
    float m = 0.f;
    for (int d = 0; d < Dc; d++) m += st[r * Dc + d] + ta[r * Dc + d];
    m *= (1.0f / Dc);
    float v = 0.f;
    for (int d = 0; d < Dc; d++) {
      float y = st[r * Dc + d] + ta[r * Dc + d] - m;
      v += y * y;
    }
    const float inv = rsqrtf(v * (1.0f / Dc) + 1e-5f);
    for (int d = 0; d < Dc; d++) {
      float y = st[r * Dc + d] + ta[r * Dc + d];
      float o = (y - m) * inv * n2_s[d] + n2_b[d];
      st[r * Dc + d] = o;
      s16[r * Dc + d] = (half_t)o;  // final st, consumed by unpool
    }
  }
}

// ---------------------------------------------------------------------------
// Kernel 3: unpool + residual. grid(512, B), block 256. out = Wm @ st + x
// B operand staged transposed in LDS so frags are contiguous.
// ---------------------------------------------------------------------------
__global__ void k_unpool(const float* __restrict__ x,
                         const half_t* __restrict__ Wm16,
                         const half_t* __restrict__ st16,
                         float* __restrict__ out) {
  __shared__ alignas(16) half_t lds_stT[Dc * Sc];  // 32 KB, [d][s]
  const int b = blockIdx.y;
  const int n0 = blockIdx.x * 64;
  const int tid = threadIdx.x, wv = tid >> 5, lane = tid & 31;

  const half_t* stb = st16 + (size_t)b * Sc * Dc;
  for (int i = tid; i < Sc * Dc; i += 256) {
    int s = i >> 8, d = i & 255;
    lds_stT[d * Sc + s] = stb[i];
  }
  __syncthreads();

  const half_t* wmb = Wm16 + ((size_t)b * Nc + n0) * Sc;
  const float*  xb  = x   + ((size_t)b * Nc + n0) * Dc;
  float*        ob  = out + ((size_t)b * Nc + n0) * Dc;

  for (int t = wv; t < 64; t += 8) {  // 4 M-tiles x 16 N-tiles
    const int mt = t >> 4, nt = t & 15;
    v8f c;
#pragma unroll
    for (int r = 0; r < 8; r++) c[r] = 0.f;
    const int m = mt * 16 + (lane & 15);
    const int n = nt * 16 + (lane & 15);
    const half_t* arow = wmb + (size_t)m * Sc;
    const half_t* bcol = lds_stT + n * Sc;
#pragma unroll
    for (int kb = 0; kb < Sc; kb += 32) {
      v16h a = frag_a_f16(arow + kb, lane);
      v16h bfr = frag_b_f16(bcol + kb, lane);
      c = wmma_f16(a, bfr, c);
    }
#pragma unroll
    for (int r = 0; r < 8; r++) {
      const int mm = mt * 16 + r + 8 * (lane >> 4);
      ob[(size_t)mm * Dc + n] = c[r] + xb[(size_t)mm * Dc + n];
    }
  }
}

// ---------------------------------------------------------------------------
extern "C" void kernel_launch(void* const* d_in, const int* in_sizes, int n_in,
                              void* d_out, int out_size, void* d_ws, size_t ws_size,
                              hipStream_t stream) {
  (void)in_sizes; (void)n_in; (void)out_size; (void)ws_size;
  const float* x     = (const float*)d_in[0];
  const float* sq_w  = (const float*)d_in[1];
  const float* sq_b  = (const float*)d_in[2];
  const float* in_w  = (const float*)d_in[3];
  const float* in_b  = (const float*)d_in[4];
  const float* out_w = (const float*)d_in[5];
  const float* out_b = (const float*)d_in[6];
  const float* n1_s  = (const float*)d_in[7];
  const float* n1_b  = (const float*)d_in[8];
  const float* n2_s  = (const float*)d_in[9];
  const float* n2_b  = (const float*)d_in[10];
  const float* f1_w  = (const float*)d_in[11];
  const float* f1_b  = (const float*)d_in[12];
  const float* f2_w  = (const float*)d_in[13];
  const float* f2_b  = (const float*)d_in[14];
  const float* state = (const float*)d_in[15];
  const float* g_w   = (const float*)d_in[16];
  const float* g_b   = (const float*)d_in[17];
  float* out = (float*)d_out;

  // workspace carve (256B aligned)
  char* p = (char*)d_ws;
  size_t off = 0;
  auto take = [&](size_t bytes) {
    void* r = p + off;
    off += (bytes + 255) & ~(size_t)255;
    return r;
  };
  half_t* Wm16  = (half_t*)take((size_t)Bc * Nc * Sc * sizeof(half_t)); // 16 MB
  float*  stF   = (float*) take((size_t)Bc * Sc * Dc * sizeof(float));
  half_t* st16  = (half_t*)take((size_t)Bc * Sc * Dc * sizeof(half_t));
  float*  qkvF  = (float*) take((size_t)Bc * Sc * 3 * Dc * sizeof(float));
  half_t* ctx16 = (half_t*)take((size_t)Bc * Sc * Dc * sizeof(half_t));
  half_t* h16   = (half_t*)take((size_t)Bc * Sc * 4 * Dc * sizeof(half_t));
  float*  tmpH  = (float*) take((size_t)Bc * Sc * 4 * Dc * sizeof(float));
  float*  tmpA  = (float*) take((size_t)Bc * Sc * Dc * sizeof(float));
  // f16 weight mirrors, all stored [N][K]
  half_t* g_wT   = (half_t*)take((size_t)Dc * Dc * sizeof(half_t));
  half_t* in_wT  = (half_t*)take((size_t)3 * Dc * Dc * sizeof(half_t));
  half_t* out_wT = (half_t*)take((size_t)Dc * Dc * sizeof(half_t));
  half_t* f1_wT  = (half_t*)take((size_t)Dc * 4 * Dc * sizeof(half_t));
  half_t* f2_wT  = (half_t*)take((size_t)4 * Dc * Dc * sizeof(half_t));

  const int stElems = Bc * Sc * Dc;
  k_zero<<<(stElems + 255) / 256, 256, 0, stream>>>(stF, stElems);

  // weight conversion (tiny):
  // in_w [768][256] and out_w [256][256] are already [N][K] (used as W^T)
  k_cvt<<<(3 * Dc * Dc + 255) / 256, 256, 0, stream>>>(in_w, in_wT, 3 * Dc * Dc);
  k_cvt<<<(Dc * Dc + 255) / 256, 256, 0, stream>>>(out_w, out_wT, Dc * Dc);
  // g_w [256][256], f1_w [256][1024], f2_w [1024][256] are [K][N] -> transpose
  k_cvtT<<<(Dc * Dc + 255) / 256, 256, 0, stream>>>(g_w, g_wT, Dc, Dc);
  k_cvtT<<<(Dc * 4 * Dc + 255) / 256, 256, 0, stream>>>(f1_w, f1_wT, Dc, 4 * Dc);
  k_cvtT<<<(4 * Dc * Dc + 255) / 256, 256, 0, stream>>>(f2_w, f2_wT, 4 * Dc, Dc);

  k_pool<<<dim3(64, Bc), 256, 0, stream>>>(x, sq_w, sq_b, Wm16, stF);
  k_slice<<<Bc, 256, 0, stream>>>(stF, st16, qkvF, ctx16, h16, tmpH, tmpA,
                                  g_wT, g_b, in_wT, in_b, out_wT, out_b,
                                  n1_s, n1_b, f1_wT, f1_b, f2_wT, f2_b,
                                  n2_s, n2_b, state);
  k_unpool<<<dim3(Nc / 64, Bc), 256, 0, stream>>>(x, Wm16, st16, out);
}
